// ToMELocalEncoder_86818468921659
// MI455X (gfx1250) — matrix operand order repaired and checked by reading, not compiled, not run on hardware
//
#include <hip/hip_runtime.h>
#include <hip/hip_bf16.h>

// ============================================================================
// ViT-ToMe encoder for MI455X (gfx1250, wave32).
// All dense GEMMs go through v_wmma_f32_16x16x32_bf16 (f32 accumulate) with
// vectorized (b128) LDS fragment loads in contiguous k order (k = 16*(lane/16) + s for A and B) and
// double-buffered LDS staging (1 barrier / K-chunk).
// Attention uses the local-window structure (<=32 keys/query) with a
// wave32 flash-softmax loop -- WMMA would waste >80% of each tile on mask.
// ============================================================================

typedef __bf16 bf16;
typedef bf16  v16bf __attribute__((ext_vector_type(16)));
typedef bf16  v8bf  __attribute__((ext_vector_type(8)));
typedef bf16  v2bf  __attribute__((ext_vector_type(2)));
typedef float v8f   __attribute__((ext_vector_type(8)));

#define D_MODEL 768
#define NHEAD   12
#define DHEAD   64
#define T0      197
#define BATCH   64
#define WINR    16
#define TOME_R  24

// Native f32->bf16 (RNE) pair pack; clang lowers to packed cvt on gfx1250.
__device__ __forceinline__ unsigned pack2bf(float lo, float hi) {
  v2bf v;
  v[0] = (bf16)lo;
  v[1] = (bf16)hi;
  return __builtin_bit_cast(unsigned, v);
}
__device__ __forceinline__ float gelu_exact(float v) {
  return 0.5f * v * (1.0f + erff(v * 0.70710678118654752f));
}

// ---------------------------------------------------------------------------
// Generic GEMM: C[M,N] = epi( A[M,K] @ W[K,N] + bias ), optional residual.
// REQUIRES: M%64==0, N%64==0, K%32==0 (true for every call site here).
// Block = 256 threads = 8 waves; block tile 64x64; wave = 2 x (16x16) tiles.
// K staged in double-buffered LDS as bf16 in chunks of 32, laid out so each
// WMMA fragment half is one contiguous 16-byte LDS load (ds_load_b128).
// ---------------------------------------------------------------------------
#define EPI_NONE 0
#define EPI_GELU 1
#define EPI_RES  2

__launch_bounds__(256)
__global__ void gemm_wmma_bf16(const float* __restrict__ A,
                               const float* __restrict__ W,
                               const float* __restrict__ bias,
                               const float* __restrict__ res,
                               float* __restrict__ C,
                               int M, int N, int K, int epi) {
  // A tile: row-major bf16; row stride 40 bf16 = 80B (16B-aligned rows).
  __shared__ __align__(16) bf16     sA[2][64][40];
  // B tile: [n][kpair] packed dwords (bf16 pair {k even, k odd}); 80B rows.
  __shared__ __align__(16) unsigned sBq[2][64][20];

  const int tid   = threadIdx.x;
  const int lane  = tid & 31;
  const int wave  = tid >> 5;          // 0..7
  const int wm    = wave & 3;          // M tile (16 rows each)
  const int wn    = wave >> 2;         // 0/1 -> N half (2 x 16 cols)
  const int khalf = lane >> 4;         // ISA lane-half K split
  const int ml    = lane & 15;

  const int mb = blockIdx.y * 64;
  const int nb = blockIdx.x * 64;

  // staging coordinates
  const int bkp = tid >> 4;            // 0..15 : k-pair row for B
  const int bn4 = (tid & 15) << 2;     // 0..60 : 4 consecutive n for B

  v8f acc0 = {}; v8f acc1 = {};

  auto stage = [&](int buf, int kb) {
    // A tile (64x32 floats -> bf16), 2 x float4 per thread
#pragma unroll
    for (int i = 0; i < 2; ++i) {
      int qid = tid + i * 256;         // 512 quads total
      int m   = qid >> 3;
      int c4  = (qid & 7) << 2;
      float4 f = *(const float4*)&A[(size_t)(mb + m) * K + kb + c4];
      uint2 u;
      u.x = pack2bf(f.x, f.y);
      u.y = pack2bf(f.z, f.w);
      *(uint2*)&sA[buf][m][c4] = u;
    }
    // B tile (32x64 floats -> packed k-pair dwords, transposed)
    float4 r0 = *(const float4*)&W[(size_t)(kb + 2 * bkp) * N + nb + bn4];
    float4 r1 = *(const float4*)&W[(size_t)(kb + 2 * bkp + 1) * N + nb + bn4];
    sBq[buf][bn4 + 0][bkp] = pack2bf(r0.x, r1.x);
    sBq[buf][bn4 + 1][bkp] = pack2bf(r0.y, r1.y);
    sBq[buf][bn4 + 2][bkp] = pack2bf(r0.z, r1.z);
    sBq[buf][bn4 + 3][bkp] = pack2bf(r0.w, r1.w);
  };

  stage(0, 0);
  __syncthreads();

  for (int kb = 0; kb < K; kb += 32) {
    const int cur = (kb >> 5) & 1;
    // stage next chunk into the other buffer (overlaps with WMMA below)
    if (kb + 32 < K) stage(cur ^ 1, kb + 32);
    // speculative prefetch two chunks ahead (-> global_prefetch_b8)
    if (kb + 64 < K) {
      __builtin_prefetch(&A[(size_t)(mb + (tid >> 3)) * K + kb + 64], 0, 1);
      __builtin_prefetch(&W[(size_t)(kb + 64 + (tid >> 4)) * N + nb], 0, 1);
    }

    // A fragment: VGPR v holds K pair k0 = khalf*16 + v*2.
    // Pairs are contiguous in sA rows -> two b128 loads per lane.
    const bf16* ap = &sA[cur][wm * 16 + ml][khalf * 16];
    v8bf a_lo = __builtin_bit_cast(v8bf, *(const uint4*)__builtin_assume_aligned(ap, 16));
    v8bf a_hi = __builtin_bit_cast(v8bf, *(const uint4*)__builtin_assume_aligned(ap + 8, 16));
    v16bf afrag = __builtin_shufflevector(a_lo, a_hi,
        0, 1, 2, 3, 4, 5, 6, 7, 8, 9, 10, 11, 12, 13, 14, 15);

    // B fragment: VGPR v holds pair (kb2+2v, kb2+2v+1), kb2 = khalf*16
    //  -> dwords sBq[n][khalf*8 + v] : 8 consecutive dwords = two b128 loads.
    const unsigned* bp0 = &sBq[cur][wn * 32 + ml][khalf * 8];
    v8bf b0lo = __builtin_bit_cast(v8bf, *(const uint4*)__builtin_assume_aligned(bp0, 16));
    v8bf b0hi = __builtin_bit_cast(v8bf, *(const uint4*)__builtin_assume_aligned(bp0 + 4, 16));
    v16bf bfrag0 = __builtin_shufflevector(b0lo, b0hi,
        0, 1, 2, 3, 4, 5, 6, 7, 8, 9, 10, 11, 12, 13, 14, 15);
    const unsigned* bp1 = &sBq[cur][wn * 32 + 16 + ml][khalf * 8];
    v8bf b1lo = __builtin_bit_cast(v8bf, *(const uint4*)__builtin_assume_aligned(bp1, 16));
    v8bf b1hi = __builtin_bit_cast(v8bf, *(const uint4*)__builtin_assume_aligned(bp1 + 4, 16));
    v16bf bfrag1 = __builtin_shufflevector(b1lo, b1hi,
        0, 1, 2, 3, 4, 5, 6, 7, 8, 9, 10, 11, 12, 13, 14, 15);

    acc0 = __builtin_amdgcn_wmma_f32_16x16x32_bf16(false, afrag, false, bfrag0,
                                                   (short)0, acc0, false, false);
    acc1 = __builtin_amdgcn_wmma_f32_16x16x32_bf16(false, afrag, false, bfrag1,
                                                   (short)0, acc1, false, false);
    __syncthreads();   // next-buffer staging complete; cur safe to reuse in 2 iters
  }

  // epilogue: C/D layout -> M = v + 8*(lane/16), N = lane%16
  const int ng0 = nb + wn * 32 + ml;
  const int ng1 = ng0 + 16;
  const float b0 = bias ? bias[ng0] : 0.0f;
  const float b1 = bias ? bias[ng1] : 0.0f;
#pragma unroll
  for (int v = 0; v < 8; ++v) {
    int mg = mb + wm * 16 + v + (khalf << 3);
    float o0 = acc0[v] + b0;
    float o1 = acc1[v] + b1;
    if (epi == EPI_GELU) {
      o0 = gelu_exact(o0); o1 = gelu_exact(o1);
    } else if (epi == EPI_RES) {
      o0 += res[(size_t)mg * N + ng0]; o1 += res[(size_t)mg * N + ng1];
    }
    C[(size_t)mg * N + ng0] = o0;
    C[(size_t)mg * N + ng1] = o1;
  }
}

// ---------------------------------------------------------------------------
// Wave-per-row LayerNorm over D=768 (24 elems / lane, shfl reductions).
// ---------------------------------------------------------------------------
__launch_bounds__(256)
__global__ void layernorm_rows(const float* __restrict__ x,
                               const float* __restrict__ g,
                               const float* __restrict__ bb,
                               float* __restrict__ y, int rows) {
  int wid  = blockIdx.x * 8 + (threadIdx.x >> 5);
  int lane = threadIdx.x & 31;
  if (wid >= rows) return;
  const float* xr = x + (size_t)wid * D_MODEL;
  float v[24], s = 0.f, ss = 0.f;
#pragma unroll
  for (int i = 0; i < 24; ++i) {
    v[i] = xr[lane + 32 * i];
    s += v[i]; ss += v[i] * v[i];
  }
  for (int o = 16; o; o >>= 1) { s += __shfl_xor(s, o, 32); ss += __shfl_xor(ss, o, 32); }
  float mu  = s * (1.0f / D_MODEL);
  float var = ss * (1.0f / D_MODEL) - mu * mu;
  float rs  = rsqrtf(var + 1e-6f);
  float* yr = y + (size_t)wid * D_MODEL;
#pragma unroll
  for (int i = 0; i < 24; ++i) {
    int c = lane + 32 * i;
    yr[c] = (v[i] - mu) * rs * g[c] + bb[c];
  }
}

// ---------------------------------------------------------------------------
// Windowed attention: one wave32 per (b,h,i). Each lane holds 2 of 64 dims.
// Mask: (|i-j|<16) | (i==0) | (j==0). Online (flash) softmax.
// qkv layout: (B,T, [q|k|v] x H x 64)
// ---------------------------------------------------------------------------
__launch_bounds__(256)
__global__ void attn_local(const float* __restrict__ qkv,
                           float* __restrict__ out, int B, int T) {
  int wid  = blockIdx.x * 8 + (threadIdx.x >> 5);
  int lane = threadIdx.x & 31;
  if (wid >= B * NHEAD * T) return;
  int i = wid % T;
  int h = (wid / T) % NHEAD;
  int b = wid / (T * NHEAD);

  const size_t rstride = 3 * D_MODEL;
  const float* base = qkv + (size_t)b * T * rstride;
  const float* qr   = base + (size_t)i * rstride + h * DHEAD;
  float q0 = qr[lane], q1 = qr[lane + 32];

  float m = -3.4e38f, l = 0.f, a0 = 0.f, a1 = 0.f;
  auto step = [&](int j) {
    const float* kr = base + (size_t)j * rstride + D_MODEL + h * DHEAD;
    const float* vr = base + (size_t)j * rstride + 2 * D_MODEL + h * DHEAD;
    float p = q0 * kr[lane] + q1 * kr[lane + 32];
    for (int o = 16; o; o >>= 1) p += __shfl_xor(p, o, 32);
    float s  = p * 0.125f;                  // dh^-0.5 = 1/8
    float nm = fmaxf(m, s);
    float cr = __expf(m - nm);
    float w  = __expf(s - nm);
    l  = l * cr + w;
    a0 = a0 * cr + w * vr[lane];
    a1 = a1 * cr + w * vr[lane + 32];
    m = nm;
  };
  if (i == 0) {
    for (int j = 0; j < T; ++j) step(j);
  } else {
    step(0);
    int jlo = (i - WINR + 1 < 1) ? 1 : i - WINR + 1;
    int jhi = (i + WINR - 1 < T - 1) ? i + WINR - 1 : T - 1;
    for (int j = jlo; j <= jhi; ++j) step(j);
  }
  float inv = 1.0f / l;
  float* orow = out + ((size_t)(b * T + i)) * D_MODEL + h * DHEAD;
  orow[lane]      = a0 * inv;
  orow[lane + 32] = a1 * inv;
}

// ---------------------------------------------------------------------------
// Misc elementwise kernels
// ---------------------------------------------------------------------------
__global__ void im2col_patches(const float* __restrict__ x, float* __restrict__ xp) {
  size_t idx = (size_t)blockIdx.x * 256 + threadIdx.x;
  const size_t total = (size_t)BATCH * 196 * D_MODEL;
  if (idx >= total) return;
  int d = idx % D_MODEL; size_t tmp = idx / D_MODEL;
  int t = tmp % 196; int b = tmp / 196;
  int c = d >> 8, rem = d & 255;
  int py = rem >> 4, px = rem & 15;
  int gy = t / 14, gx = t % 14;
  xp[idx] = x[((size_t)(b * 3 + c) * 224 + gy * 16 + py) * 224 + gx * 16 + px];
}

__global__ void assemble_tok(const float* __restrict__ pt, const float* __restrict__ cls,
                             const float* __restrict__ pos, float* __restrict__ tok) {
  size_t idx = (size_t)blockIdx.x * 256 + threadIdx.x;
  const size_t total = (size_t)BATCH * T0 * D_MODEL;
  if (idx >= total) return;
  int d = idx % D_MODEL; size_t tmp = idx / D_MODEL;
  int t = tmp % T0; int b = tmp / T0;
  float v = (t == 0) ? cls[d] : pt[((size_t)b * 196 + (t - 1)) * D_MODEL + d];
  tok[idx] = v + pos[(size_t)t * D_MODEL + d];
}

__global__ void copy_f32(const float* __restrict__ s, float* __restrict__ d, size_t n) {
  size_t i = (size_t)blockIdx.x * 256 + threadIdx.x;
  if (i < n) d[i] = s[i];
}
__global__ void fill_f32(float* __restrict__ d, float v, size_t n) {
  size_t i = (size_t)blockIdx.x * 256 + threadIdx.x;
  if (i < n) d[i] = v;
}
__global__ void iota_smap(int* __restrict__ sm) {
  int i = blockIdx.x * 256 + threadIdx.x;
  if (i < BATCH * T0) sm[i] = i % T0;
}
__global__ void copy_size_out(const float* __restrict__ sz, float* __restrict__ o, int tn) {
  int i = blockIdx.x * 256 + threadIdx.x;
  if (i < BATCH * tn) o[i] = sz[(i / tn) * T0 + (i % tn)];
}
__global__ void copy_smap_out(const int* __restrict__ s, int* __restrict__ d) {
  int i = blockIdx.x * 256 + threadIdx.x;
  if (i < BATCH * T0) d[i] = s[i];
}

// ---------------------------------------------------------------------------
// ToMe: L2-normalize rows (wave per row)
// ---------------------------------------------------------------------------
__launch_bounds__(256)
__global__ void l2norm_rows(const float* __restrict__ x, float* __restrict__ y, int rows) {
  int wid  = blockIdx.x * 8 + (threadIdx.x >> 5);
  int lane = threadIdx.x & 31;
  if (wid >= rows) return;
  const float* xr = x + (size_t)wid * D_MODEL;
  float v[24], ss = 0.f;
#pragma unroll
  for (int i = 0; i < 24; ++i) { v[i] = xr[lane + 32 * i]; ss += v[i] * v[i]; }
  for (int o = 16; o; o >>= 1) ss += __shfl_xor(ss, o, 32);
  float inv = rsqrtf(ss);
  float* yr = y + (size_t)wid * D_MODEL;
#pragma unroll
  for (int i = 0; i < 24; ++i) yr[lane + 32 * i] = v[i] * inv;
}

// Per-row max/argmax of scores (wave per (b,i)); tie -> smallest j (jnp.argmax)
__launch_bounds__(256)
__global__ void tome_scores(const float* __restrict__ Mn, float* __restrict__ nmax,
                            int* __restrict__ nidx, int t, int t1, int t2) {
  int wid  = blockIdx.x * 8 + (threadIdx.x >> 5);
  int lane = threadIdx.x & 31;
  if (wid >= BATCH * t1) return;
  int i = wid % t1, b = wid / t1;
  if (i == 0) {               // scores.at[:,0,:] = -inf
    if (lane == 0) { nmax[b * 128] = -3.4e38f; nidx[b * 128] = 0; }
    return;
  }
  const float* ai = Mn + ((size_t)b * t + 2 * i) * D_MODEL;
  float best = -3.4e38f; int bj = 0x7fffffff;
  for (int j = lane; j < t2; j += 32) {
    const float* bjp = Mn + ((size_t)b * t + 2 * j + 1) * D_MODEL;
    float d = 0.f;
    for (int c = 0; c < D_MODEL; ++c) d += ai[c] * bjp[c];
    if (d > best) { best = d; bj = j; }
  }
  for (int o = 16; o; o >>= 1) {
    float ob = __shfl_xor(best, o, 32);
    int   oj = __shfl_xor(bj, o, 32);
    if (ob > best || (ob == best && oj < bj)) { best = ob; bj = oj; }
  }
  if (lane == 0) { nmax[b * 128 + i] = best; nidx[b * 128 + i] = bj; }
}

// One thread per batch: stable descending argsort (matches jnp.argsort(-x)),
// build unm/src/dst, and update the source map in place.
__global__ void tome_plan(const float* __restrict__ nmax, const int* __restrict__ nidx,
                          int* __restrict__ unm, int* __restrict__ src, int* __restrict__ dst,
                          int* __restrict__ smap, int t, int t1, int t2) {
  int b = blockIdx.x * blockDim.x + threadIdx.x;
  if (b >= BATCH) return;
  float nm[104]; int ni[104]; unsigned char used[104]; int edge[104];
  for (int i = 0; i < t1; ++i) {
    nm[i] = nmax[b * 128 + i]; ni[i] = nidx[b * 128 + i]; used[i] = 0;
  }
  for (int pos = 0; pos < t1; ++pos) {          // stable selection (strict >)
    float bv = -__builtin_inff(); int bi = 0;
    for (int i = 0; i < t1; ++i)
      if (!used[i] && nm[i] > bv) { bv = nm[i]; bi = i; }
    used[bi] = 1; edge[pos] = bi;
  }
  const int ul = t1 - TOME_R;
  int un[104];
  for (int q = 0; q < ul; ++q) un[q] = edge[TOME_R + q];
  for (int q = 1; q < ul; ++q) {                 // ascending insertion sort
    int key = un[q], p = q - 1;
    while (p >= 0 && un[p] > key) { un[p + 1] = un[p]; --p; }
    un[p + 1] = key;
  }
  int sc[TOME_R], dc[TOME_R];
  for (int k = 0; k < TOME_R; ++k) { sc[k] = edge[k]; dc[k] = ni[sc[k]]; }
  for (int q = 0; q < ul; ++q) unm[b * 104 + q] = un[q];
  for (int k = 0; k < TOME_R; ++k) { src[b * 32 + k] = sc[k]; dst[b * 32 + k] = dc[k]; }
  // old -> new token index map, applied to source_map
  int e[104];
  for (int i = 0; i < t1; ++i) e[i] = 0;
  for (int q = 0; q < ul; ++q) e[un[q]] = q;
  for (int k = 0; k < TOME_R; ++k) e[sc[k]] = ul + dc[k];
  int o2n[200];
  for (int i = 0; i < t1; ++i) o2n[2 * i] = e[i];
  for (int j = 0; j < t2; ++j) o2n[2 * j + 1] = ul + j;
  for (int s = 0; s < T0; ++s) smap[b * T0 + s] = o2n[smap[b * T0 + s]];
}

// Merge: init (gather unm + dst rows, scaled by size), scatter-add src, divide.
__global__ void tome_merge_init(const float* __restrict__ x, const float* __restrict__ sz,
                                const int* __restrict__ unm, float* __restrict__ xs,
                                float* __restrict__ szn, int t, int t1, int t2) {
  const int ul = t1 - TOME_R, tn = ul + t2;
  size_t idx = (size_t)blockIdx.x * 256 + threadIdx.x;
  size_t total = (size_t)BATCH * tn * D_MODEL;
  if (idx >= total) return;
  int d = idx % D_MODEL; size_t tmp = idx / D_MODEL;
  int o = tmp % tn; int b = tmp / tn;
  int row = (o < ul) ? 2 * unm[b * 104 + o] : 2 * (o - ul) + 1;
  float s = sz[b * T0 + row];
  xs[idx] = x[((size_t)b * t + row) * D_MODEL + d] * s;
  if (d == 0) szn[b * T0 + o] = s;
}

__global__ void tome_merge_scatter(const float* __restrict__ x, const float* __restrict__ sz,
                                   const int* __restrict__ src, const int* __restrict__ dst,
                                   float* __restrict__ xs, float* __restrict__ szn,
                                   int t, int t1, int t2) {
  const int ul = t1 - TOME_R, tn = ul + t2;
  size_t idx = (size_t)blockIdx.x * 256 + threadIdx.x;
  size_t total = (size_t)BATCH * TOME_R * D_MODEL;
  if (idx >= total) return;
  int d = idx % D_MODEL; size_t tmp = idx / D_MODEL;
  int k = tmp % TOME_R; int b = tmp / TOME_R;
  int row = 2 * src[b * 32 + k];
  int o   = ul + dst[b * 32 + k];
  float s = sz[b * T0 + row];
  atomicAdd(&xs[((size_t)b * tn + o) * D_MODEL + d],
            x[((size_t)b * t + row) * D_MODEL + d] * s);
  if (d == 0) atomicAdd(&szn[b * T0 + o], s);
}

__global__ void tome_merge_div(float* __restrict__ xs, const float* __restrict__ szn, int tn) {
  size_t idx = (size_t)blockIdx.x * 256 + threadIdx.x;
  size_t total = (size_t)BATCH * tn * D_MODEL;
  if (idx >= total) return;
  size_t tmp = idx / D_MODEL;
  int o = tmp % tn; int b = tmp / tn;
  xs[idx] /= szn[b * T0 + o];
}

// ===========================================================================
// Host orchestration
// ===========================================================================
static inline size_t cdiv(size_t a, size_t b) { return (a + b - 1) / b; }

extern "C" void kernel_launch(void* const* d_in, const int* in_sizes, int n_in,
                              void* d_out, int out_size, void* d_ws, size_t ws_size,
                              hipStream_t stream) {
  (void)in_sizes; (void)n_in; (void)out_size; (void)ws_size;

  // -------- inputs (JAX pytree leaf order: 'x', then params sorted keys:
  // blocks[0..3]x{fc1_b,fc1_w,fc2_b,fc2_w,ln1_b,ln1_g,ln2_b,ln2_g,proj_b,
  // proj_w,qkv_b,qkv_w}, cls, norm_b, norm_g, patch_b, patch_w, pos) --------
  const float* x = (const float*)d_in[0];
  struct Blk { const float *fc1_b,*fc1_w,*fc2_b,*fc2_w,*ln1_b,*ln1_g,*ln2_b,*ln2_g,
                           *proj_b,*proj_w,*qkv_b,*qkv_w; };
  Blk blk[4];
  int p = 1;
  for (int i = 0; i < 4; ++i) {
    blk[i].fc1_b  = (const float*)d_in[p++]; blk[i].fc1_w  = (const float*)d_in[p++];
    blk[i].fc2_b  = (const float*)d_in[p++]; blk[i].fc2_w  = (const float*)d_in[p++];
    blk[i].ln1_b  = (const float*)d_in[p++]; blk[i].ln1_g  = (const float*)d_in[p++];
    blk[i].ln2_b  = (const float*)d_in[p++]; blk[i].ln2_g  = (const float*)d_in[p++];
    blk[i].proj_b = (const float*)d_in[p++]; blk[i].proj_w = (const float*)d_in[p++];
    blk[i].qkv_b  = (const float*)d_in[p++]; blk[i].qkv_w  = (const float*)d_in[p++];
  }
  const float* cls     = (const float*)d_in[p++];
  const float* norm_b  = (const float*)d_in[p++];
  const float* norm_g  = (const float*)d_in[p++];
  const float* patch_b = (const float*)d_in[p++];
  const float* patch_w = (const float*)d_in[p++];
  const float* pos     = (const float*)d_in[p++];

  // -------- workspace layout (floats) --------
  const size_t BT  = (size_t)BATCH * T0;            // 12608
  const size_t SZX = BT * D_MODEL;                  // 9,682,944
  float* base = (float*)d_ws;
  float* X    = base;                               // tokens / residual stream
  float* Hb   = X   + SZX;                          // LN output / ToMe metric
  float* QKV  = Hb  + SZX;                          // (B,T,2304); also im2col
  float* ATT  = QKV + BT * 3 * D_MODEL;             // attn out; ToMe ping A
  float* MLP  = ATT + SZX;                          // (B,T,3072); also patch tmp
  float* PB   = MLP + BT * 4 * D_MODEL;             // ToMe ping B
  float* sizeA = PB + SZX;                          // 64*197
  float* sizeB = sizeA + BT;
  float* nmax  = sizeB + BT;                        // 64*128
  int* nidx = (int*)(nmax + BATCH * 128);           // 64*128
  int* unmI = nidx + BATCH * 128;                   // 64*104
  int* srcI = unmI + BATCH * 104;                   // 64*32
  int* dstI = srcI + BATCH * 32;                    // 64*32
  int* smap = dstI + BATCH * 32;                    // 64*197

  // -------- output layout (x_out, x_embed, size, source_map concat) --------
  float* out_x     = (float*)d_out;                            // (64,101,768)
  float* out_embed = out_x + (size_t)BATCH * 101 * D_MODEL;    // (64,197,768)
  float* out_sz    = out_embed + SZX;                          // (64,101,1)
  int*   out_smap  = (int*)(out_sz + (size_t)BATCH * 101);     // (64,197)

  dim3 blk256(256);

  // ======== 1. Patch embedding ========
  {
    float* xp = QKV;   // im2col scratch (free at this point)
    float* pt = MLP;   // patch GEMM output (B,196,768)
    im2col_patches<<<cdiv((size_t)BATCH * 196 * D_MODEL, 256), blk256, 0, stream>>>(x, xp);
    dim3 g(D_MODEL / 64, (BATCH * 196) / 64);
    gemm_wmma_bf16<<<g, blk256, 0, stream>>>(xp, patch_w, patch_b, nullptr, pt,
                                             BATCH * 196, D_MODEL, 3 * 16 * 16, EPI_NONE);
    assemble_tok<<<cdiv(SZX, 256), blk256, 0, stream>>>(pt, cls, pos, X);
  }

  // ======== 2. Transformer blocks ========
  for (int l = 0; l < 4; ++l) {
    layernorm_rows<<<cdiv(BT, 8), blk256, 0, stream>>>(X, blk[l].ln1_g, blk[l].ln1_b, Hb, (int)BT);
    {
      dim3 g((3 * D_MODEL) / 64, BT / 64);
      gemm_wmma_bf16<<<g, blk256, 0, stream>>>(Hb, blk[l].qkv_w, blk[l].qkv_b, nullptr, QKV,
                                               (int)BT, 3 * D_MODEL, D_MODEL, EPI_NONE);
    }
    attn_local<<<cdiv((size_t)BATCH * NHEAD * T0, 8), blk256, 0, stream>>>(QKV, ATT, BATCH, T0);
    {
      dim3 g(D_MODEL / 64, BT / 64);
      gemm_wmma_bf16<<<g, blk256, 0, stream>>>(ATT, blk[l].proj_w, blk[l].proj_b, X, X,
                                               (int)BT, D_MODEL, D_MODEL, EPI_RES);
    }
    layernorm_rows<<<cdiv(BT, 8), blk256, 0, stream>>>(X, blk[l].ln2_g, blk[l].ln2_b, Hb, (int)BT);
    {
      dim3 g((4 * D_MODEL) / 64, BT / 64);
      gemm_wmma_bf16<<<g, blk256, 0, stream>>>(Hb, blk[l].fc1_w, blk[l].fc1_b, nullptr, MLP,
                                               (int)BT, 4 * D_MODEL, D_MODEL, EPI_GELU);
    }
    {
      dim3 g(D_MODEL / 64, BT / 64);
      gemm_wmma_bf16<<<g, blk256, 0, stream>>>(MLP, blk[l].fc2_w, blk[l].fc2_b, X, X,
                                               (int)BT, D_MODEL, 4 * D_MODEL, EPI_RES);
    }
  }

  // x_embed out
  copy_f32<<<cdiv(SZX, 256), blk256, 0, stream>>>(X, out_embed, SZX);

  // ======== 3. ToMe merging (4 iterations) ========
  fill_f32<<<cdiv(BT, 256), blk256, 0, stream>>>(sizeA, 1.0f, BT);
  iota_smap<<<cdiv(BT, 256), blk256, 0, stream>>>(smap);

  float* xping[2] = {ATT, PB};
  float* sping[2] = {sizeB, sizeA};
  int t = T0;
  for (int it = 0; it < 4; ++it) {
    const int t1 = (t + 1) / 2, t2 = t / 2, tn = t - TOME_R;
    float* xin  = (it == 0) ? X : xping[(it + 1) & 1];
    float* xout = xping[it & 1];
    float* sin  = (it == 0) ? sizeA : sping[(it + 1) & 1];
    float* sout = sping[it & 1];

    l2norm_rows<<<cdiv((size_t)BATCH * t, 8), blk256, 0, stream>>>(xin, Hb, BATCH * t);
    tome_scores<<<cdiv((size_t)BATCH * t1, 8), blk256, 0, stream>>>(Hb, nmax, nidx, t, t1, t2);
    tome_plan<<<1, 64, 0, stream>>>(nmax, nidx, unmI, srcI, dstI, smap, t, t1, t2);
    tome_merge_init<<<cdiv((size_t)BATCH * tn * D_MODEL, 256), blk256, 0, stream>>>(
        xin, sin, unmI, xout, sout, t, t1, t2);
    tome_merge_scatter<<<cdiv((size_t)BATCH * TOME_R * D_MODEL, 256), blk256, 0, stream>>>(
        xin, sin, srcI, dstI, xout, sout, t, t1, t2);
    tome_merge_div<<<cdiv((size_t)BATCH * tn * D_MODEL, 256), blk256, 0, stream>>>(xout, sout, tn);
    t = tn;
  }
  float* xfin = xping[1];    // after 4 iters (T=101)
  float* sfin = sping[1];

  // ======== 4. Final LN + remaining outputs ========
  layernorm_rows<<<cdiv((size_t)BATCH * t, 8), blk256, 0, stream>>>(xfin, norm_g, norm_b,
                                                                    out_x, BATCH * t);
  copy_size_out<<<cdiv((size_t)BATCH * t, 256), blk256, 0, stream>>>(sfin, out_sz, t);
  copy_smap_out<<<cdiv(BT, 256), blk256, 0, stream>>>(smap, out_smap);
}
